// MaskedSelfAttention_31636729102763
// MI455X (gfx1250) — matrix operand order repaired and checked by reading, not compiled, not run on hardware
//
#include <hip/hip_runtime.h>
#include <hip/hip_bf16.h>

// ---------------------------------------------------------------------------
// Problem dims (fixed by the reference): B=4, S=2048, D=1024, H=16, DK=DV=64
// ---------------------------------------------------------------------------
constexpr int CB = 4;
constexpr int CS = 2048;
constexpr int CD = 1024;
constexpr int CH = 16;
constexpr int CM = CB * CS;          // 8192 rows

// ---------------------------------------------------------------------------
// Types for CDNA5 WMMA (wave32): D(16x16 f32) = A(16x32 bf16) x B(32x16 bf16) + C
// ---------------------------------------------------------------------------
typedef __attribute__((ext_vector_type(16))) __bf16 v16bf;
typedef __attribute__((ext_vector_type(8)))  float  v8f;

union B16x16 {            // 16 bf16 elements == 32 bytes == two b128 chunks
    v16bf v;
    uint4 u[2];
};

static __device__ __forceinline__ v8f wmma_bf16(v16bf a, v16bf b, v8f c) {
    return __builtin_amdgcn_wmma_f32_16x16x32_bf16(
        /*neg_a=*/false, a, /*neg_b=*/false, b,
        /*c_mod=*/(short)0, c, /*reuse_a=*/false, /*reuse_b=*/false);
}

static __device__ __forceinline__ unsigned short f2bf(float f) {
    unsigned int u = __float_as_uint(f);
    unsigned int r = u + 0x7FFFu + ((u >> 16) & 1u);   // round-to-nearest-even
    return (unsigned short)(r >> 16);
}

// ---------------------------------------------------------------------------
// Kernel 1: f32 -> bf16 conversion, 4 elements/thread (all sizes % 4 == 0)
// ---------------------------------------------------------------------------
__global__ __launch_bounds__(256) void cvt_f32_bf16_k(
    const float* __restrict__ in, unsigned short* __restrict__ out, long n) {
    long i = ((long)blockIdx.x * blockDim.x + threadIdx.x) * 4;
    if (i + 3 < n) {
        float4 f = *(const float4*)(in + i);
        unsigned int p0 = (unsigned int)f2bf(f.x) | ((unsigned int)f2bf(f.y) << 16);
        unsigned int p1 = (unsigned int)f2bf(f.z) | ((unsigned int)f2bf(f.w) << 16);
        uint2 o; o.x = p0; o.y = p1;
        *(uint2*)(out + i) = o;
    }
}

// ---------------------------------------------------------------------------
// Kernel 2: bf16 WMMA GEMM  Y = X @ W + bias   (X:[8192][1024], W:[1024][1024])
//   MODE 0: write bf16 into [B][H][S][64] head-major (QKV projections)
//   MODE 1: write f32 row-major [8192][1024] (output projection -> d_out)
// Block: 256 thr (8 waves), tile 128x128, wave tile 32(M) x 64(N), K-step 32.
// ---------------------------------------------------------------------------
#define GEMM_TM 128
#define GEMM_TN 128
#define GEMM_TK 32
#define LDA_S   40   // elems; row pitch 80B (16B aligned)
#define LDB_S   40

template <int MODE>
__global__ __launch_bounds__(256) void gemm_bf16_k(
    const unsigned short* __restrict__ A,
    const unsigned short* __restrict__ W,
    const float* __restrict__ bias,
    void* __restrict__ Out) {

    constexpr int N = CD;   // 1024
    constexpr int K = CD;   // 1024

    __shared__ __align__(16) unsigned short lA[GEMM_TM * LDA_S];      // [m][k]
    __shared__ __align__(16) unsigned short lB[GEMM_TN * LDB_S];      // [n][k] (transposed)

    const int tid  = threadIdx.x;
    const int lane = tid & 31;
    const int wid  = tid >> 5;
    const int l16  = lane & 15;
    const int half = lane >> 4;          // 0: lanes 0-15, 1: lanes 16-31
    const int wave_m = wid >> 1;         // 0..3  (32 rows each)
    const int wave_n = wid & 1;          // 0..1  (64 cols each)
    const int m0 = blockIdx.x * GEMM_TM;
    const int n0 = blockIdx.y * GEMM_TN;

    v8f acc[2][4];
    #pragma unroll
    for (int mi = 0; mi < 2; ++mi)
        #pragma unroll
        for (int ni = 0; ni < 4; ++ni)
            acc[mi][ni] = (v8f){0.f,0.f,0.f,0.f,0.f,0.f,0.f,0.f};

    for (int kb = 0; kb < K; kb += GEMM_TK) {
        // --- stage A tile 128x32 (row-major) ---
        {
            int row = tid >> 1;
            int kp  = (tid & 1) * 16;
            const unsigned short* src = A + (size_t)(m0 + row) * K + kb + kp;
            uint4 d0 = *(const uint4*)(src);
            uint4 d1 = *(const uint4*)(src + 8);
            *(uint4*)(&lA[row * LDA_S + kp])     = d0;
            *(uint4*)(&lA[row * LDA_S + kp + 8]) = d1;
        }
        // --- stage B tile 32x128 transposed into [n][k] ---
        {
            int k  = tid & 31;
            int nc = tid >> 5;           // 0..7, 16 cols each
            const unsigned short* src = W + (size_t)(kb + k) * N + n0 + nc * 16;
            uint4 d0 = *(const uint4*)(src);
            uint4 d1 = *(const uint4*)(src + 8);
            const unsigned short* e0 = (const unsigned short*)&d0;
            const unsigned short* e1 = (const unsigned short*)&d1;
            #pragma unroll
            for (int i = 0; i < 8; ++i) lB[(nc*16 + i)     * LDB_S + k] = e0[i];
            #pragma unroll
            for (int i = 0; i < 8; ++i) lB[(nc*16 + 8 + i) * LDB_S + k] = e1[i];
        }
        __syncthreads();

        // --- fragments ---
        v16bf afr[2], bfr[4];
        #pragma unroll
        for (int mi = 0; mi < 2; ++mi) {
            int row = wave_m * 32 + mi * 16 + l16;
            int kA  = half * 8;          // lo lanes: K0..7 & 16..23; hi: 8..15 & 24..31
            B16x16 t;
            t.u[0] = *(const uint4*)(&lA[row * LDA_S + kA]);
            t.u[1] = *(const uint4*)(&lA[row * LDA_S + kA + 16]);
            afr[mi] = t.v;
        }
        #pragma unroll
        for (int ni = 0; ni < 4; ++ni) {
            int col = wave_n * 64 + ni * 16 + l16;
            int kB  = half * 8;          // lo lanes: K0..7 & 16..23; hi: 8..15 & 24..31
            B16x16 t;
            t.u[0] = *(const uint4*)(&lB[col * LDB_S + kB]);
            t.u[1] = *(const uint4*)(&lB[col * LDB_S + kB + 16]);
            bfr[ni] = t.v;
        }
        #pragma unroll
        for (int mi = 0; mi < 2; ++mi)
            #pragma unroll
            for (int ni = 0; ni < 4; ++ni)
                acc[mi][ni] = wmma_bf16(afr[mi], bfr[ni], acc[mi][ni]);
        __syncthreads();
    }

    // --- epilogue (all address math is shifts/masks: dims are powers of two) ---
    #pragma unroll
    for (int mi = 0; mi < 2; ++mi) {
        #pragma unroll
        for (int ni = 0; ni < 4; ++ni) {
            int col = n0 + wave_n * 64 + ni * 16 + l16;
            float bv = bias[col];
            #pragma unroll
            for (int v = 0; v < 8; ++v) {
                int row = m0 + wave_m * 32 + mi * 16 + v + half * 8;
                float val = acc[mi][ni][v] + bv;
                if constexpr (MODE == 0) {
                    int bb = row >> 11;            // row / CS  (CS = 2048)
                    int s  = row & (CS - 1);
                    int h  = col >> 6;
                    int dk = col & 63;
                    size_t off = ((((size_t)bb * CH + h) * CS + s) << 6) + dk;
                    ((unsigned short*)Out)[off] = f2bf(val);
                } else {
                    ((float*)Out)[(size_t)row * N + col] = val;
                }
            }
        }
    }
}

// ---------------------------------------------------------------------------
// Kernel 3: causal flash attention.
//   Q,K,V: bf16 [B][H][S][64].  O: bf16 [B][S][H*64].
//   Block = 256 thr (8 waves) handles 128 query rows of one (b,h).
//   Wave owns 16 q-rows; KV streamed in 64-row tiles with online softmax.
// ---------------------------------------------------------------------------
#define AQ  128
#define AKV 64
#define AST 72      // elems; row pitch 144B (16B aligned)

__global__ __launch_bounds__(256) void flash_attn_k(
    const unsigned short* __restrict__ Q,
    const unsigned short* __restrict__ Km,
    const unsigned short* __restrict__ V,
    unsigned short* __restrict__ O) {

    __shared__ __align__(16) unsigned short lQ [AQ  * AST];        // [q][dk]
    __shared__ __align__(16) unsigned short lK [AKV * AST];        // [kv][dk] == B-frag [n][k]
    __shared__ __align__(16) unsigned short lVt[AKV * AST];        // [dv][kv]
    __shared__ __align__(16) unsigned short lP [8 * 16 * AST];     // per-wave P tiles

    const int tid  = threadIdx.x;
    const int lane = tid & 31;
    const int wid  = tid >> 5;
    const int l16  = lane & 15;
    const int half = lane >> 4;

    constexpr int NQ = CS / AQ;          // 16 query blocks per (b,h)
    const int bx   = blockIdx.x;
    const int qblk = bx & (NQ - 1);
    const int h    = (bx >> 4) & (CH - 1);
    const int b    = bx >> 8;
    const int qbase = qblk * AQ;

    const size_t headOff = (((size_t)b * CH + h) * CS) * 64;
    const unsigned short* Qh = Q  + headOff;
    const unsigned short* Kh = Km + headOff;
    const unsigned short* Vh = V  + headOff;

    // stage Q tile: 128 rows x 64 cols
    {
        int row  = tid >> 1;
        int part = (tid & 1) * 32;
        const unsigned short* src = Qh + (size_t)(qbase + row) * 64 + part;
        uint4 d0 = ((const uint4*)src)[0];
        uint4 d1 = ((const uint4*)src)[1];
        uint4 d2 = ((const uint4*)src)[2];
        uint4 d3 = ((const uint4*)src)[3];
        uint4* dst = (uint4*)&lQ[row * AST + part];
        dst[0] = d0; dst[1] = d1; dst[2] = d2; dst[3] = d3;
    }

    v8f oacc[4];
    #pragma unroll
    for (int d = 0; d < 4; ++d) oacc[d] = (v8f){0.f,0.f,0.f,0.f,0.f,0.f,0.f,0.f};
    float mrun[8], lrun[8];
    #pragma unroll
    for (int v = 0; v < 8; ++v) { mrun[v] = -3.0e38f; lrun[v] = 0.f; }

    const int qrow0 = wid * 16;
    const int jend  = 2 * qblk + 2;     // kv tiles of 64 needed for causal coverage
    unsigned short* Pw = &lP[wid * 16 * AST];

    for (int j = 0; j < jend; ++j) {
        __syncthreads();   // prior-iter LDS readers done (also covers Q staging)
        // stage K tile [kv][dk]
        {
            int row  = tid >> 2;
            int part = (tid & 3) * 16;
            const unsigned short* src = Kh + (size_t)(j * 64 + row) * 64 + part;
            uint4 d0 = ((const uint4*)src)[0];
            uint4 d1 = ((const uint4*)src)[1];
            uint4* dst = (uint4*)&lK[row * AST + part];
            dst[0] = d0; dst[1] = d1;
        }
        // stage V transposed -> lVt[dv][kv]
        {
            int kv   = tid >> 2;
            int part = (tid & 3) * 16;
            const unsigned short* src = Vh + (size_t)(j * 64 + kv) * 64 + part;
            uint4 d0 = ((const uint4*)src)[0];
            uint4 d1 = ((const uint4*)src)[1];
            const unsigned short* e0 = (const unsigned short*)&d0;
            const unsigned short* e1 = (const unsigned short*)&d1;
            #pragma unroll
            for (int i = 0; i < 8; ++i) lVt[(part + i)     * AST + kv] = e0[i];
            #pragma unroll
            for (int i = 0; i < 8; ++i) lVt[(part + 8 + i) * AST + kv] = e1[i];
        }
        __syncthreads();

        // S = Q @ K^T  (16 x 64 per wave)
        v8f sacc[4];
        #pragma unroll
        for (int nf = 0; nf < 4; ++nf) sacc[nf] = (v8f){0.f,0.f,0.f,0.f,0.f,0.f,0.f,0.f};
        #pragma unroll
        for (int ks = 0; ks < 2; ++ks) {
            int kA = ks * 32 + half * 8;
            B16x16 ta;
            ta.u[0] = *(const uint4*)&lQ[(qrow0 + l16) * AST + kA];
            ta.u[1] = *(const uint4*)&lQ[(qrow0 + l16) * AST + kA + 16];
            #pragma unroll
            for (int nf = 0; nf < 4; ++nf) {
                int col = nf * 16 + l16;
                int kB  = ks * 32 + half * 8;
                B16x16 tb;
                tb.u[0] = *(const uint4*)&lK[col * AST + kB];
                tb.u[1] = *(const uint4*)&lK[col * AST + kB + 16];
                sacc[nf] = wmma_bf16(ta.v, tb.v, sacc[nf]);
            }
        }

        // scale + causal mask + online softmax
        float pv[4][8], rmax[8], alpha[8];
        #pragma unroll
        for (int v = 0; v < 8; ++v) {
            int qg = qbase + qrow0 + v + half * 8;
            float mx = -3.0e38f;
            #pragma unroll
            for (int nf = 0; nf < 4; ++nf) {
                int kg = j * 64 + nf * 16 + l16;
                float sv = sacc[nf][v] * 0.125f;          // 1/sqrt(64)
                if (kg > qg) sv = -3.0e38f;
                pv[nf][v] = sv;
                mx = fmaxf(mx, sv);
            }
            rmax[v] = mx;
        }
        #pragma unroll
        for (int v = 0; v < 8; ++v) {
            float mx = rmax[v];
            mx = fmaxf(mx, __shfl_xor(mx, 1, 32));
            mx = fmaxf(mx, __shfl_xor(mx, 2, 32));
            mx = fmaxf(mx, __shfl_xor(mx, 4, 32));
            mx = fmaxf(mx, __shfl_xor(mx, 8, 32));
            float mnew = fmaxf(mrun[v], mx);
            alpha[v] = __expf(mrun[v] - mnew);
            mrun[v]  = mnew;
        }
        #pragma unroll
        for (int v = 0; v < 8; ++v) {
            float sum = 0.f;
            #pragma unroll
            for (int nf = 0; nf < 4; ++nf) {
                float p = __expf(pv[nf][v] - mrun[v]);
                pv[nf][v] = p;
                sum += p;
            }
            sum += __shfl_xor(sum, 1, 32);
            sum += __shfl_xor(sum, 2, 32);
            sum += __shfl_xor(sum, 4, 32);
            sum += __shfl_xor(sum, 8, 32);
            lrun[v] = lrun[v] * alpha[v] + sum;
            #pragma unroll
            for (int df = 0; df < 4; ++df) oacc[df][v] *= alpha[v];
        }

        // P (accumulator layout) -> LDS (A-fragment layout source)
        #pragma unroll
        for (int nf = 0; nf < 4; ++nf)
            #pragma unroll
            for (int v = 0; v < 8; ++v)
                Pw[(v + half * 8) * AST + nf * 16 + l16] = f2bf(pv[nf][v]);
        __syncthreads();

        // O += P @ V
        #pragma unroll
        for (int ks = 0; ks < 2; ++ks) {
            int kA = ks * 32 + half * 8;
            B16x16 ta;
            ta.u[0] = *(const uint4*)&Pw[l16 * AST + kA];
            ta.u[1] = *(const uint4*)&Pw[l16 * AST + kA + 16];
            #pragma unroll
            for (int df = 0; df < 4; ++df) {
                int drow = df * 16 + l16;
                int kB   = ks * 32 + half * 8;
                B16x16 tb;
                tb.u[0] = *(const uint4*)&lVt[drow * AST + kB];
                tb.u[1] = *(const uint4*)&lVt[drow * AST + kB + 16];
                oacc[df] = wmma_bf16(ta.v, tb.v, oacc[df]);
            }
        }
    }

    // normalize and store O: [b][s][h*64 + dv]
    #pragma unroll
    for (int df = 0; df < 4; ++df) {
        int dv = df * 16 + l16;
        #pragma unroll
        for (int v = 0; v < 8; ++v) {
            int srow = qbase + qrow0 + v + half * 8;
            float val = oacc[df][v] / lrun[v];
            size_t off = ((size_t)b * CS + srow) * ((size_t)CH * 64) + (size_t)h * 64 + dv;
            O[off] = f2bf(val);
        }
    }
}

// ---------------------------------------------------------------------------
// Host launcher
// ---------------------------------------------------------------------------
extern "C" void kernel_launch(void* const* d_in, const int* in_sizes, int n_in,
                              void* d_out, int out_size, void* d_ws, size_t ws_size,
                              hipStream_t stream) {
    (void)in_sizes; (void)n_in; (void)out_size; (void)ws_size;

    const float* x  = (const float*)d_in[0];
    const float* Wq = (const float*)d_in[1];
    const float* bq = (const float*)d_in[2];
    const float* Wk = (const float*)d_in[3];
    const float* bk = (const float*)d_in[4];
    const float* Wv = (const float*)d_in[5];
    const float* bv = (const float*)d_in[6];
    const float* Wo = (const float*)d_in[7];
    const float* bo = (const float*)d_in[8];
    float* out = (float*)d_out;

    const size_t xElems = (size_t)CM * CD;      // 8.39M
    const size_t wElems = (size_t)CD * CD;      // 1.05M

    char* ws = (char*)d_ws;
    unsigned short* xb  = (unsigned short*)ws; ws += xElems * 2;
    unsigned short* Wqb = (unsigned short*)ws; ws += wElems * 2;
    unsigned short* Wkb = (unsigned short*)ws; ws += wElems * 2;
    unsigned short* Wvb = (unsigned short*)ws; ws += wElems * 2;
    unsigned short* Wob = (unsigned short*)ws; ws += wElems * 2;
    unsigned short* Qb  = (unsigned short*)ws; ws += xElems * 2;
    unsigned short* Kb  = (unsigned short*)ws; ws += xElems * 2;
    unsigned short* Vb  = (unsigned short*)ws; ws += xElems * 2;
    unsigned short* Ob  = (unsigned short*)ws; ws += xElems * 2;

    // converts
    cvt_f32_bf16_k<<<dim3((unsigned)(xElems / 1024)), 256, 0, stream>>>(x,  xb,  (long)xElems);
    cvt_f32_bf16_k<<<dim3((unsigned)(wElems / 1024)), 256, 0, stream>>>(Wq, Wqb, (long)wElems);
    cvt_f32_bf16_k<<<dim3((unsigned)(wElems / 1024)), 256, 0, stream>>>(Wk, Wkb, (long)wElems);
    cvt_f32_bf16_k<<<dim3((unsigned)(wElems / 1024)), 256, 0, stream>>>(Wv, Wvb, (long)wElems);
    cvt_f32_bf16_k<<<dim3((unsigned)(wElems / 1024)), 256, 0, stream>>>(Wo, Wob, (long)wElems);

    dim3 ggrid((unsigned)(CM / GEMM_TM), (unsigned)(CD / GEMM_TN));  // 64 x 8
    gemm_bf16_k<0><<<ggrid, 256, 0, stream>>>(xb, Wqb, bq, Qb);
    gemm_bf16_k<0><<<ggrid, 256, 0, stream>>>(xb, Wkb, bk, Kb);
    gemm_bf16_k<0><<<ggrid, 256, 0, stream>>>(xb, Wvb, bv, Vb);

    flash_attn_k<<<dim3((unsigned)(CB * CH * (CS / AQ))), 256, 0, stream>>>(Qb, Kb, Vb, Ob);

    gemm_bf16_k<1><<<ggrid, 256, 0, stream>>>(Ob, Wob, bo, out);
}
